// TSAE_77060303225531
// MI455X (gfx1250) — matrix operand order repaired and checked by reading, not compiled, not run on hardware
//
#include <hip/hip_runtime.h>
#include <cstdint>
#include <cstddef>

// ---------------------------------------------------------------------------
// TSAE for MI455X (gfx1250, wave32, WMMA).
// Heavy GEMMs: v_wmma_f32_16x16x32_bf16, 128x128 tiles, 8 wmma/wave/K-step,
// double-buffered GLOBAL_LOAD_ASYNC_TO_LDS_B128 staging (ASYNCcnt overlap).
// Attention-1: full 64x512 score strip resident in LDS (320KB/WGP).
// ---------------------------------------------------------------------------

typedef __attribute__((ext_vector_type(16))) __bf16 v16bf;
typedef __attribute__((ext_vector_type(8)))  __bf16 v8bf;
typedef __attribute__((ext_vector_type(8)))  float  v8f;

constexpr int cB = 4, cD = 12, cL = 512, cH = 768, cNH = 12, cM = 4096, cTOPK = 64, cHD = 64;
constexpr int NROW = cB * cD * cL;            // 24576 rows of H
constexpr size_t NELT = (size_t)NROW * cH;    // 18,874,368 elements

// ---- async global->LDS copy (CDNA5, ASYNCcnt) ------------------------------
// VDST VGPR = LDS byte address (low 32 bits of a generic LDS pointer ARE the
// LDS offset on amdgcn).
#define USE_ASYNC_LDS 1

__device__ __forceinline__ void g2lds_b128(const void* g, void* l)
{
#if USE_ASYNC_LDS
    asm volatile("global_load_async_to_lds_b128 %0, %1, off"
                 :: "v"((unsigned)(unsigned long long)l),
                    "v"((unsigned long long)g)
                 : "memory");
#else
    *(v8bf*)l = *(const v8bf*)g;
#endif
}

__device__ __forceinline__ void wait_async()
{
#if USE_ASYNC_LDS
    asm volatile("s_wait_asynccnt 0x0" ::: "memory");
#endif
}

// ---------------------------------------------------------------------------
// WMMA fragment K-mapping per ISA 7.12.2 (bf16):
//   A (16x32): lane group g = lane>>4 ; element e : K = (e<8 ? e : e+8) + 8*g
//   B (32x16): lane n = lane&15, g = lane>>4 : element e : K as for A (this code loads both A and B with K = 16*g + e)
//   C/D 16x16 f32: VGPR r, lane: M = r + 8*(lane>>4), N = lane&15
// ---------------------------------------------------------------------------

// ======================= generic tiled WMMA GEMM ===========================
// C = A @ W^T (+bias)(+relu)(+res). A [Mrows,K], W [Ncols,K], bf16 row-major.
// Block = 256 threads (8 waves), tile 128x128, K-step 32, ping-pong staging:
// iteration i computes from buffer (i&1) while async copies fill the other.
template <int BIAS, int RELU, int RES, int OUTF, int OUTB>
__launch_bounds__(256)
__global__ void gemm_wmma(const __bf16* __restrict__ A, const __bf16* __restrict__ W,
                          const float* __restrict__ bias, const float* __restrict__ res,
                          float* __restrict__ Cf, __bf16* __restrict__ Ch,
                          int Kdim, int Ncols)
{
    __shared__ __bf16 As[2][128 * 32];   // [buf][m][k]
    __shared__ __bf16 Bs[2][128 * 32];   // [buf][n][k]
    const int m0  = blockIdx.x * 128;
    const int n0  = blockIdx.y * 128;
    const int tid = threadIdx.x;
    const int w   = tid >> 5, lane = tid & 31;
    const int mb  = (w & 3) * 32;     // wave M base in tile
    const int nb  = (w >> 2) * 64;    // wave N base in tile
    const int g   = lane >> 4, ln16 = lane & 15;

    auto stage = [&](int buf, int kk) {
        #pragma unroll
        for (int half = 0; half < 2; ++half) {
            int idx = tid * 16 + half * 8;
            int r = idx >> 5, c = idx & 31;
            g2lds_b128(&A[(size_t)(m0 + r) * Kdim + kk + c], &As[buf][idx]);
            g2lds_b128(&W[(size_t)(n0 + r) * Kdim + kk + c], &Bs[buf][idx]);
        }
    };

    v8f acc[2][4] = {};

    stage(0, 0);                 // prologue: fill buffer 0
    wait_async();
    __syncthreads();

    for (int k0 = 0; k0 < Kdim; k0 += 32) {
        const int cur = (k0 >> 5) & 1;
        if (k0 + 32 < Kdim) stage(cur ^ 1, k0 + 32);   // overlap with compute
        if (k0 + 64 < Kdim) {                          // prefetch 2 tiles ahead
            int idx = tid * 16;
            int r = idx >> 5, c = idx & 31;
            __builtin_prefetch(&A[(size_t)(m0 + r) * Kdim + k0 + 64 + c], 0, 3);
            __builtin_prefetch(&W[(size_t)(n0 + r) * Kdim + k0 + 64 + c], 0, 3);
        }

        v16bf a[2], b[4];
        #pragma unroll
        for (int e = 0; e < 16; ++e) {
            int ka = g * 16 + e;
            a[0][e] = As[cur][(mb +      ln16) * 32 + ka];
            a[1][e] = As[cur][(mb + 16 + ln16) * 32 + ka];
            int kb = g * 16 + e;
            b[0][e] = Bs[cur][(nb +      ln16) * 32 + kb];
            b[1][e] = Bs[cur][(nb + 16 + ln16) * 32 + kb];
            b[2][e] = Bs[cur][(nb + 32 + ln16) * 32 + kb];
            b[3][e] = Bs[cur][(nb + 48 + ln16) * 32 + kb];
        }
        #pragma unroll
        for (int mi = 0; mi < 2; ++mi)
            #pragma unroll
            for (int ni = 0; ni < 4; ++ni)
                acc[mi][ni] = __builtin_amdgcn_wmma_f32_16x16x32_bf16(
                    false, a[mi], false, b[ni], (short)0, acc[mi][ni], false, false);

        wait_async();            // next buffer ready, current reads done
        __syncthreads();
    }

    #pragma unroll
    for (int mi = 0; mi < 2; ++mi)
        #pragma unroll
        for (int ni = 0; ni < 4; ++ni)
            #pragma unroll
            for (int r = 0; r < 8; ++r) {
                int row = m0 + mb + mi * 16 + r + 8 * g;
                int col = n0 + nb + ni * 16 + ln16;
                float v = acc[mi][ni][r];
                if (BIAS) v += bias[col];
                if (RELU) v = fmaxf(v, 0.f);
                if (RES)  v += res[(size_t)row * Ncols + col];
                if (OUTF) Cf[(size_t)row * Ncols + col] = v;
                if (OUTB) Ch[(size_t)row * Ncols + col] = (__bf16)v;
            }
}

// ================= fused attention over L (non-causal, key mask) ===========
// One block = 64 query rows of one (bd, head). Full 64x512 score strip in LDS.
// Waves 4(M) x 2(N): each wave 16x32 of the 64x64 S / O tile.
// K/V tiles are ping-pong staged with async copies.
__launch_bounds__(256)
__global__ void attn_seq_kernel(const __bf16* __restrict__ Q, const __bf16* __restrict__ K,
                                const __bf16* __restrict__ V, const unsigned char* __restrict__ mask,
                                __bf16* __restrict__ O)
{
    extern __shared__ char smem[];
    float*  S    = (float*)smem;                                  // 64*512 f32 (128KB)
    __bf16* P    = (__bf16*)(smem + 64 * 512 * 4);                // 64*512 bf16 (64KB)
    __bf16* Qs   = (__bf16*)(smem + 64 * 512 * 4 + 64 * 512 * 2); // 64*64 bf16
    __bf16* KVs  = Qs + 64 * 64;                                  // 2 x 64*64 bf16
    float*  madd = (float*)(KVs + 2 * 64 * 64);                   // 512 f32 additive mask
    __shared__ float red[256];

    const int qb = blockIdx.x, head = blockIdx.y, bd = blockIdx.z;
    const int bb  = bd / cD;
    const int tid = threadIdx.x;
    const int w = tid >> 5, lane = tid & 31;
    const int wm = w & 3, wn = w >> 2;
    const int g = lane >> 4, ln16 = lane & 15;
    const int q0 = qb * 64;

    auto stageKV = [&](const __bf16* src, int buf, int kb2) {
        #pragma unroll
        for (int half = 0; half < 2; ++half) {
            int idx = tid * 16 + half * 8;
            int r = idx >> 6, c = idx & 63;
            g2lds_b128(&src[((size_t)(bd * cL + kb2 * 64 + r)) * cH + head * cHD + c],
                       &KVs[buf * 4096 + idx]);
        }
    };

    // prologue: Q tile, first K tile, additive key mask
    #pragma unroll
    for (int half = 0; half < 2; ++half) {
        int idx = tid * 16 + half * 8;
        int r = idx >> 6, c = idx & 63;
        g2lds_b128(&Q[((size_t)(bd * cL + q0 + r)) * cH + head * cHD + c], &Qs[idx]);
    }
    stageKV(K, 0, 0);
    for (int i = tid; i < cL; i += 256) madd[i] = mask[bb * cL + i] ? 0.f : -1e9f;
    wait_async();
    __syncthreads();

    // ---- S = Q @ K^T * scale + mask ----
    for (int kb = 0; kb < cL / 64; ++kb) {
        const int cur = kb & 1;
        if (kb + 1 < cL / 64) stageKV(K, cur ^ 1, kb + 1);
        const __bf16* kv = KVs + cur * 4096;
        v8f s0 = {}, s1 = {};
        #pragma unroll
        for (int k0 = 0; k0 < 64; k0 += 32) {
            v16bf a, b0, b1;
            #pragma unroll
            for (int e = 0; e < 16; ++e) {
                int ka = k0 + g * 16 + e;
                a[e]  = Qs[(wm * 16 + ln16) * 64 + ka];
                int kk = k0 + g * 16 + e;
                b0[e] = kv[(wn * 32 + ln16) * 64 + kk];       // B[k][n] = K[n][k]
                b1[e] = kv[(wn * 32 + 16 + ln16) * 64 + kk];
            }
            s0 = __builtin_amdgcn_wmma_f32_16x16x32_bf16(false, a, false, b0, (short)0, s0, false, false);
            s1 = __builtin_amdgcn_wmma_f32_16x16x32_bf16(false, a, false, b1, (short)0, s1, false, false);
        }
        #pragma unroll
        for (int t = 0; t < 2; ++t) {
            int kc = kb * 64 + wn * 32 + t * 16 + ln16;
            float ma = madd[kc];                               // hoisted, one LDS read
            #pragma unroll
            for (int r = 0; r < 8; ++r) {
                int qr = wm * 16 + r + 8 * g;
                S[qr * cL + kc] = (t ? s1[r] : s0[r]) * 0.125f + ma;   // 1/sqrt(64)
            }
        }
        wait_async();
        __syncthreads();
    }

    // ---- exact softmax, 4 threads per row ----
    {
        const int rrow = tid >> 2, part = tid & 3;
        const int j0 = part * (cL / 4), j1 = j0 + cL / 4;
        float mx = -3.0e38f;
        for (int j = j0; j < j1; ++j) mx = fmaxf(mx, S[rrow * cL + j]);
        red[tid] = mx; __syncthreads();
        float gmx = fmaxf(fmaxf(red[rrow * 4], red[rrow * 4 + 1]),
                          fmaxf(red[rrow * 4 + 2], red[rrow * 4 + 3]));
        __syncthreads();
        float sum = 0.f;
        for (int j = j0; j < j1; ++j) {
            float e = __expf(S[rrow * cL + j] - gmx);
            S[rrow * cL + j] = e; sum += e;
        }
        red[tid] = sum; __syncthreads();
        float inv = 1.f / (red[rrow * 4] + red[rrow * 4 + 1] + red[rrow * 4 + 2] + red[rrow * 4 + 3]);
        for (int j = j0; j < j1; ++j) P[rrow * cL + j] = (__bf16)(S[rrow * cL + j] * inv);
    }
    __syncthreads();

    // ---- O = P @ V ----
    v8f o0 = {}, o1 = {};
    stageKV(V, 0, 0);
    wait_async();
    __syncthreads();
    for (int kb = 0; kb < cL / 64; ++kb) {
        const int cur = kb & 1;
        if (kb + 1 < cL / 64) stageKV(V, cur ^ 1, kb + 1);
        const __bf16* kv = KVs + cur * 4096;
        #pragma unroll
        for (int k0 = 0; k0 < 64; k0 += 32) {
            v16bf a, b0, b1;
            #pragma unroll
            for (int e = 0; e < 16; ++e) {
                int ka = kb * 64 + k0 + g * 16 + e;
                a[e]  = P[(wm * 16 + ln16) * cL + ka];
                int kr = k0 + g * 16 + e;
                b0[e] = kv[kr * 64 + wn * 32 + ln16];         // B[k][n] = V[k][n]
                b1[e] = kv[kr * 64 + wn * 32 + 16 + ln16];
            }
            o0 = __builtin_amdgcn_wmma_f32_16x16x32_bf16(false, a, false, b0, (short)0, o0, false, false);
            o1 = __builtin_amdgcn_wmma_f32_16x16x32_bf16(false, a, false, b1, (short)0, o1, false, false);
        }
        wait_async();
        __syncthreads();
    }
    #pragma unroll
    for (int t = 0; t < 2; ++t)
        #pragma unroll
        for (int r = 0; r < 8; ++r) {
            int qr = wm * 16 + r + 8 * g;
            int oc = wn * 32 + t * 16 + ln16;
            O[((size_t)(bd * cL + q0 + qr)) * cH + head * cHD + oc] = (__bf16)(t ? o1[r] : o0[r]);
        }
}

constexpr int ATTN_SMEM = 64 * 512 * 4 + 64 * 512 * 2 + 3 * 64 * 64 * 2 + 512 * 4; // 223232 B

// ================= tiny causal attention over depth (D=12) ================
// 12x12 scores: WMMA tiles would be mostly padding -> VALU scalar path.
__launch_bounds__(256)
__global__ void depth_attn_kernel(const __bf16* __restrict__ Q, const __bf16* __restrict__ K,
                                  const __bf16* __restrict__ V, __bf16* __restrict__ O)
{
    const int bl = blockIdx.x;           // b*L + l
    const int bb = bl / cL, ll = bl % cL;
    const int t  = threadIdx.x;
    if (t >= cNH * cD) return;
    const int head = t / cD, dq = t % cD;
    const size_t rq = ((size_t)(bb * cD + dq) * cL + ll) * cH + head * cHD;

    float q[cHD];
    #pragma unroll
    for (int c = 0; c < cHD; ++c) q[c] = (float)Q[rq + c];

    float sc[cD];
    float mx = -3.0e38f;
    for (int dk = 0; dk <= dq; ++dk) {
        size_t rk = ((size_t)(bb * cD + dk) * cL + ll) * cH + head * cHD;
        float s = 0.f;
        #pragma unroll
        for (int c = 0; c < cHD; ++c) s += q[c] * (float)K[rk + c];
        s *= 0.125f;
        sc[dk] = s; mx = fmaxf(mx, s);
    }
    float sum = 0.f;
    for (int dk = 0; dk <= dq; ++dk) { sc[dk] = __expf(sc[dk] - mx); sum += sc[dk]; }
    float inv = 1.f / sum;

    float o[cHD];
    #pragma unroll
    for (int c = 0; c < cHD; ++c) o[c] = 0.f;
    for (int dk = 0; dk <= dq; ++dk) {
        size_t rv = ((size_t)(bb * cD + dk) * cL + ll) * cH + head * cHD;
        float pp = sc[dk] * inv;
        #pragma unroll
        for (int c = 0; c < cHD; ++c) o[c] += pp * (float)V[rv + c];
    }
    #pragma unroll
    for (int c = 0; c < cHD; ++c) O[rq + c] = (__bf16)o[c];
}

// ============================ small kernels ===============================
__global__ void f32_to_bf16_kernel(const float* __restrict__ in, __bf16* __restrict__ out, size_t n)
{
    size_t i = (size_t)blockIdx.x * blockDim.x + threadIdx.x;
    if (i < n) out[i] = (__bf16)in[i];
}

// x[b,d,l,h] = d==0 ? query_token[h] : zL[b,d-1,l,h]
__global__ void build_x_kernel(const float* __restrict__ zL, const float* __restrict__ qt,
                               float* __restrict__ x)
{
    size_t i = (size_t)blockIdx.x * blockDim.x + threadIdx.x;
    if (i >= NELT) return;
    int h = (int)(i % cH);
    size_t row = i / cH;
    int l = (int)(row % cL);
    size_t bd = row / cL;
    int d = (int)(bd % cD);
    int b = (int)(bd / cD);
    x[i] = (d == 0) ? qt[h] : zL[(((size_t)(b * cD + d - 1)) * cL + l) * cH + h];
}

__launch_bounds__(256)
__global__ void layernorm_bf16_kernel(const float* __restrict__ x, const float* __restrict__ gam,
                                      const float* __restrict__ bet, __bf16* __restrict__ y)
{
    __shared__ float red[256];
    const int row = blockIdx.x, tid = threadIdx.x;
    const float* xr = x + (size_t)row * cH;
    float s = 0.f;
    for (int i = tid; i < cH; i += 256) s += xr[i];
    red[tid] = s; __syncthreads();
    for (int o = 128; o > 0; o >>= 1) { if (tid < o) red[tid] += red[tid + o]; __syncthreads(); }
    float mu = red[0] / cH; __syncthreads();
    float v = 0.f;
    for (int i = tid; i < cH; i += 256) { float d = xr[i] - mu; v += d * d; }
    red[tid] = v; __syncthreads();
    for (int o = 128; o > 0; o >>= 1) { if (tid < o) red[tid] += red[tid + o]; __syncthreads(); }
    float rstd = rsqrtf(red[0] / cH + 1e-5f);
    for (int i = tid; i < cH; i += 256)
        y[(size_t)row * cH + i] = (__bf16)((xr[i] - mu) * rstd * gam[i] + bet[i]);
}

// A_enc = bf16( zL - x_prior - bias_pre )
__global__ void prep_enc_kernel(const float* __restrict__ zL, const float* __restrict__ xp,
                                const float* __restrict__ bias_pre, __bf16* __restrict__ out)
{
    size_t i = (size_t)blockIdx.x * blockDim.x + threadIdx.x;
    if (i >= NELT) return;
    int h = (int)(i % cH);
    out[i] = (__bf16)(zL[i] - xp[i] - bias_pre[h]);
}

// top-64 of 4096 per row: 64 iterative argmax reductions over an LDS copy
__launch_bounds__(256)
__global__ void topk_kernel(const float* __restrict__ logits, float* __restrict__ zv, int* __restrict__ zi)
{
    __shared__ float vals[cM];
    __shared__ float rv[256];
    __shared__ int   ri[256];
    const int row = blockIdx.x, tid = threadIdx.x;
    for (int i = tid; i < cM; i += 256) vals[i] = logits[(size_t)row * cM + i];
    __syncthreads();
    for (int j = 0; j < cTOPK; ++j) {
        float bm = -3.0e38f; int bi = 0;
        for (int i = tid; i < cM; i += 256)
            if (vals[i] > bm) { bm = vals[i]; bi = i; }
        rv[tid] = bm; ri[tid] = bi; __syncthreads();
        for (int o = 128; o > 0; o >>= 1) {
            if (tid < o && rv[tid + o] > rv[tid]) { rv[tid] = rv[tid + o]; ri[tid] = ri[tid + o]; }
            __syncthreads();
        }
        if (tid == 0) {
            zv[(size_t)row * cTOPK + j] = rv[0];
            zi[(size_t)row * cTOPK + j] = ri[0];
            vals[ri[0]] = -3.0e38f;
        }
        __syncthreads();
    }
}

// x_tgt[n,h] = bias_pre[h] + sum_j zv[n,j] * dict_dec[h, zi[n,j]]   (64-nnz sparse)
__launch_bounds__(256)
__global__ void decode_kernel(const float* __restrict__ zv, const int* __restrict__ zi,
                              const float* __restrict__ dict_dec, const float* __restrict__ bias_pre,
                              float* __restrict__ xt)
{
    __shared__ float sv[cTOPK];
    __shared__ int   si[cTOPK];
    const int row = blockIdx.x, tid = threadIdx.x;
    if (tid < cTOPK) { sv[tid] = zv[(size_t)row * cTOPK + tid]; si[tid] = zi[(size_t)row * cTOPK + tid]; }
    __syncthreads();
    for (int h = tid; h < cH; h += 256) {
        float a = bias_pre[h];
        #pragma unroll 8
        for (int j = 0; j < cTOPK; ++j) a += sv[j] * dict_dec[(size_t)h * cM + si[j]];
        xt[(size_t)row * cH + h] = a;
    }
}

// masked SSE of (xp - zL)  -> acc[0]
__launch_bounds__(256)
__global__ void loss_attn_kernel(const float* __restrict__ xp, const float* __restrict__ zL,
                                 const unsigned char* __restrict__ mask, float* __restrict__ acc)
{
    __shared__ float red[256];
    float s = 0.f;
    for (size_t i = (size_t)blockIdx.x * blockDim.x + threadIdx.x; i < NELT;
         i += (size_t)gridDim.x * blockDim.x) {
        size_t row = i / cH;
        int l = (int)(row % cL);
        int b = (int)(row / ((size_t)cD * cL));
        float m = mask[b * cL + l] ? 1.f : 0.f;
        float d = xp[i] - zL[i];
        s += d * d * m;
    }
    red[threadIdx.x] = s; __syncthreads();
    for (int o = 128; o > 0; o >>= 1) { if (threadIdx.x < o) red[threadIdx.x] += red[threadIdx.x + o]; __syncthreads(); }
    if (threadIdx.x == 0) atomicAdd(&acc[0], red[0]);
}

// masked SSE of (x_tgt - (zL - x_prior)) -> acc[1]
__launch_bounds__(256)
__global__ void loss_sae_kernel(const float* __restrict__ xt, const float* __restrict__ zL,
                                const float* __restrict__ xp, const unsigned char* __restrict__ mask,
                                float* __restrict__ acc)
{
    __shared__ float red[256];
    float s = 0.f;
    for (size_t i = (size_t)blockIdx.x * blockDim.x + threadIdx.x; i < NELT;
         i += (size_t)gridDim.x * blockDim.x) {
        size_t row = i / cH;
        int l = (int)(row % cL);
        int b = (int)(row / ((size_t)cD * cL));
        float m = mask[b * cL + l] ? 1.f : 0.f;
        float d = xt[i] - (zL[i] - xp[i]);
        s += d * d * m;
    }
    red[threadIdx.x] = s; __syncthreads();
    for (int o = 128; o > 0; o >>= 1) { if (threadIdx.x < o) red[threadIdx.x] += red[threadIdx.x + o]; __syncthreads(); }
    if (threadIdx.x == 0) atomicAdd(&acc[1], red[0]);
}

__launch_bounds__(256)
__global__ void mask_sum_kernel(const unsigned char* __restrict__ mask, float* __restrict__ acc)
{
    __shared__ float red[256];
    float s = 0.f;
    for (int i = threadIdx.x; i < cB * cL; i += 256) s += mask[i] ? 1.f : 0.f;
    red[threadIdx.x] = s; __syncthreads();
    for (int o = 128; o > 0; o >>= 1) { if (threadIdx.x < o) red[threadIdx.x] += red[threadIdx.x + o]; __syncthreads(); }
    if (threadIdx.x == 0) acc[2] = red[0];
}

__global__ void finalize_kernel(const float* __restrict__ acc, float* __restrict__ out)
{
    if (threadIdx.x == 0 && blockIdx.x == 0) {
        float nv = acc[2] * (float)cD * (float)cH;
        if (nv < 1.f) nv = 1.f;
        out[0] = (acc[0] + acc[1]) / nv;
    }
}

// ============================== launcher ==================================
extern "C" void kernel_launch(void* const* d_in, const int* in_sizes, int n_in,
                              void* d_out, int out_size, void* d_ws, size_t ws_size,
                              hipStream_t stream)
{
    (void)in_sizes; (void)n_in; (void)out_size; (void)ws_size;

    const float*         zL      = (const float*)d_in[0];
    const unsigned char* mask    = (const unsigned char*)d_in[1]; // numpy bool = 1 byte
    const float* Wq_l = (const float*)d_in[2];
    const float* Wk_l = (const float*)d_in[3];
    const float* Wv_l = (const float*)d_in[4];
    const float* Wo_l = (const float*)d_in[5];
    const float* g_l  = (const float*)d_in[6];
    const float* b_l  = (const float*)d_in[7];
    const float* Wq_d = (const float*)d_in[8];
    const float* Wk_d = (const float*)d_in[9];
    const float* Wv_d = (const float*)d_in[10];
    const float* Wo_d = (const float*)d_in[11];
    const float* g_d  = (const float*)d_in[12];
    const float* b_d  = (const float*)d_in[13];
    const float* dict_enc = (const float*)d_in[14]; // [M,H]
    const float* dict_dec = (const float*)d_in[15]; // [H,M]
    const float* bias_pre = (const float*)d_in[16];
    const float* bias_enc = (const float*)d_in[17];
    const float* q_tok    = (const float*)d_in[18];

    float* out     = (float*)d_out;
    float* x_prior = out + 1;               // [B,D,L,H]
    float* x_tgt   = out + 1 + NELT;        // [B,D,L,H]

    // ---- workspace bump allocator (~770 MB total) ----
    char* p = (char*)d_ws;
    auto alloc = [&](size_t bytes) { char* r = p; p += (bytes + 255) & ~(size_t)255; return r; };
    const size_t HH = (size_t)cH * cH;

    float*  acc   = (float*)alloc(64);
    __bf16* wq1   = (__bf16*)alloc(HH * 2);
    __bf16* wk1   = (__bf16*)alloc(HH * 2);
    __bf16* wv1   = (__bf16*)alloc(HH * 2);
    __bf16* wo1   = (__bf16*)alloc(HH * 2);
    __bf16* wq2   = (__bf16*)alloc(HH * 2);
    __bf16* wk2   = (__bf16*)alloc(HH * 2);
    __bf16* wv2   = (__bf16*)alloc(HH * 2);
    __bf16* wo2   = (__bf16*)alloc(HH * 2);
    __bf16* wenc  = (__bf16*)alloc((size_t)cM * cH * 2);
    float*  xbuf  = (float*)alloc(NELT * 4);       // x (with query token)
    __bf16* lnb   = (__bf16*)alloc(NELT * 2);      // LN output / SAE encoder input
    __bf16* Qb    = (__bf16*)alloc(NELT * 2);
    __bf16* Kb    = (__bf16*)alloc(NELT * 2);
    __bf16* Vb    = (__bf16*)alloc(NELT * 2);
    __bf16* attnb = (__bf16*)alloc(NELT * 2);      // pre-Wo attention output
    float*  x1    = (float*)alloc(NELT * 4);       // after attention-1 residual
    float*  logits= (float*)alloc((size_t)NROW * cM * 4);
    float*  zv    = (float*)alloc((size_t)NROW * cTOPK * 4);
    int*    zi    = (int*)alloc((size_t)NROW * cTOPK * 4);

    (void)hipMemsetAsync(acc, 0, 64, stream);

    // ---- weights -> bf16 ----
    const int CB = 256;
    auto cvt = [&](const float* src, __bf16* dst, size_t n) {
        f32_to_bf16_kernel<<<(unsigned)((n + CB - 1) / CB), CB, 0, stream>>>(src, dst, n);
    };
    cvt(Wq_l, wq1, HH); cvt(Wk_l, wk1, HH); cvt(Wv_l, wv1, HH); cvt(Wo_l, wo1, HH);
    cvt(Wq_d, wq2, HH); cvt(Wk_d, wk2, HH); cvt(Wv_d, wv2, HH); cvt(Wo_d, wo2, HH);
    cvt(dict_enc, wenc, (size_t)cM * cH);

    const unsigned EW = (unsigned)(NELT / CB);     // elementwise grid (exact)

    // ---- stage 1: build x, LN, QKV, attention over L, Wo + residual ----
    build_x_kernel<<<EW, CB, 0, stream>>>(zL, q_tok, xbuf);
    layernorm_bf16_kernel<<<NROW, CB, 0, stream>>>(xbuf, g_l, b_l, lnb);

    dim3 gproj(NROW / 128, cH / 128);
    gemm_wmma<0,0,0,0,1><<<gproj, 256, 0, stream>>>(lnb, wq1, nullptr, nullptr, nullptr, Qb, cH, cH);
    gemm_wmma<0,0,0,0,1><<<gproj, 256, 0, stream>>>(lnb, wk1, nullptr, nullptr, nullptr, Kb, cH, cH);
    gemm_wmma<0,0,0,0,1><<<gproj, 256, 0, stream>>>(lnb, wv1, nullptr, nullptr, nullptr, Vb, cH, cH);

    (void)hipFuncSetAttribute((const void*)attn_seq_kernel,
                              hipFuncAttributeMaxDynamicSharedMemorySize, ATTN_SMEM);
    dim3 gattn(cL / 64, cNH, cB * cD);
    attn_seq_kernel<<<gattn, 256, ATTN_SMEM, stream>>>(Qb, Kb, Vb, mask, attnb);

    gemm_wmma<0,0,1,1,0><<<gproj, 256, 0, stream>>>(attnb, wo1, nullptr, xbuf, x1, nullptr, cH, cH);

    // ---- stage 2: LN, QKV, causal attention over D, Wo + residual -> x_prior ----
    layernorm_bf16_kernel<<<NROW, CB, 0, stream>>>(x1, g_d, b_d, lnb);
    gemm_wmma<0,0,0,0,1><<<gproj, 256, 0, stream>>>(lnb, wq2, nullptr, nullptr, nullptr, Qb, cH, cH);
    gemm_wmma<0,0,0,0,1><<<gproj, 256, 0, stream>>>(lnb, wk2, nullptr, nullptr, nullptr, Kb, cH, cH);
    gemm_wmma<0,0,0,0,1><<<gproj, 256, 0, stream>>>(lnb, wv2, nullptr, nullptr, nullptr, Vb, cH, cH);

    depth_attn_kernel<<<cB * cL, 256, 0, stream>>>(Qb, Kb, Vb, attnb);

    gemm_wmma<0,0,1,1,0><<<gproj, 256, 0, stream>>>(attnb, wo2, nullptr, x1, x_prior, nullptr, cH, cH);

    // ---- attention reconstruction loss + mask count ----
    loss_attn_kernel<<<2048, CB, 0, stream>>>(x_prior, zL, mask, acc);
    mask_sum_kernel<<<1, CB, 0, stream>>>(mask, acc);

    // ---- SAE: encode (GEMM + bias + relu), top-k, sparse decode ----
    prep_enc_kernel<<<EW, CB, 0, stream>>>(zL, x_prior, bias_pre, lnb);
    dim3 genc(NROW / 128, cM / 128);
    gemm_wmma<1,1,0,1,0><<<genc, 256, 0, stream>>>(lnb, wenc, bias_enc, nullptr, logits, nullptr, cH, cM);

    topk_kernel<<<NROW, CB, 0, stream>>>(logits, zv, zi);
    decode_kernel<<<NROW, CB, 0, stream>>>(zv, zi, dict_dec, bias_pre, x_tgt);

    loss_sae_kernel<<<2048, CB, 0, stream>>>(x_tgt, zL, x_prior, mask, acc);
    finalize_kernel<<<1, 64, 0, stream>>>(acc, out);
}
